// MS_Block_Conv_STAtten_79216376807546
// MI455X (gfx1250) — matrix operand order repaired and checked by reading, not compiled, not run on hardware
//
#include <hip/hip_runtime.h>
#include <hip/hip_bf16.h>

typedef __attribute__((ext_vector_type(16))) _Float16 v16h;
typedef __attribute__((ext_vector_type(8)))  float    v8f;

#define TT  4
#define BB  8
#define CC  512
#define VV  1024
#define HH  8
#define DD  64
#define LL  2048   // CHUNK * V
#define NC  2      // T / CHUNK
#define HID 2048

#define LAY_PLAIN 0
#define LAY_QV    1
#define LAY_KT    2

// ---- WMMA fragment loaders (wave32 layouts per CDNA5 ISA 7.12.2) ----

__device__ __forceinline__ int kA_off(int half, int i) {
  // A (16x32 f16): lanes 0-15 hold K 0..15; lanes 16-31 hold K 16..31
  // (contiguous, matching the B fragments).
  return half * 16 + i;
}

__device__ __forceinline__ v16h load_a_rowmajor(const _Float16* __restrict__ A,
                                                int m, int lda, int k0, int half) {
  v16h a;
#pragma unroll
  for (int i = 0; i < 16; ++i) a[i] = A[(size_t)m * lda + k0 + kA_off(half, i)];
  return a;
}

// B (32x16 f16): lane = column N holds K = half*16 + i. With K-major storage the
// 16 halfs are contiguous -> two b128 loads.
__device__ __forceinline__ v16h load_row16(const _Float16* __restrict__ src) {
  v16h r;
#pragma unroll
  for (int i = 0; i < 16; ++i) r[i] = src[i];
  return r;
}

// Async global->LDS copies (CDNA5, ASYNCcnt-tracked).
__device__ __forceinline__ void wait_async0() {
  asm volatile("s_wait_asynccnt 0x0" ::: "memory");
}

// Stage a 64(N) x 32(K) f16 tile from K-major global storage into LDS buffer
// `buf` (assumes the tile array is the kernel's only LDS object, at offset 0).
// 128 threads x 2 x 16B = 4KB.
__device__ __forceinline__ void stage_btile(const _Float16* __restrict__ XTtb,
                                            int Ci, int n0, int k0, int tid, int buf) {
#pragma unroll
  for (int s0 = 0; s0 < 2; ++s0) {
    int s   = tid + s0 * 128;   // 0..255
    int row = s >> 2;           // 0..63 (N)
    int seg = s & 3;            // 16B segment within the 64B row
    const _Float16* gp = XTtb + (size_t)(n0 + row) * Ci + k0 + seg * 8;
    unsigned lb = (unsigned)(buf * 4096 + row * 64 + seg * 16);
    asm volatile("global_load_async_to_lds_b128 %0, %1, off"
                 :: "v"(lb), "v"(gp) : "memory");
  }
}

// ---- elementwise kernels ----

__global__ void cvt_f16_kernel(const float* __restrict__ X, _Float16* __restrict__ Y, int n) {
  int i = blockIdx.x * blockDim.x + threadIdx.x;
  if (i < n) Y[i] = (_Float16)X[i];
}

// LIF over T on the C-major input x, emitting K-major (T,B,V,C) f16 spikes.
__global__ void lif_x_tr_kernel(const float* __restrict__ X, _Float16* __restrict__ S) {
  const size_t plane = (size_t)BB * CC * VV;
  size_t i = (size_t)blockIdx.x * blockDim.x + threadIdx.x;
  if (i >= plane) return;
  int v = (int)(i % VV);
  int c = (int)((i / VV) % CC);
  int b = (int)(i / ((size_t)VV * CC));
  size_t o = ((size_t)b * VV + v) * CC + c;
  float vm = 0.0f;
#pragma unroll
  for (int t = 0; t < TT; ++t) {
    float u    = X[t * plane + i];
    float hpot = 0.5f * (vm + u);
    bool  sp   = (hpot >= 1.0f);
    S[t * plane + o] = sp ? (_Float16)1.0f : (_Float16)0.0f;
    vm = sp ? 0.0f : hpot;
  }
}

// Layout-preserving LIF over T (input and output share the same layout).
__global__ void lif_lin_kernel(const float* __restrict__ X, _Float16* __restrict__ S,
                               size_t plane, float th) {
  size_t i = (size_t)blockIdx.x * blockDim.x + threadIdx.x;
  if (i >= plane) return;
  float vm = 0.0f;
#pragma unroll
  for (int t = 0; t < TT; ++t) {
    float u    = X[(size_t)t * plane + i];
    float hpot = 0.5f * (vm + u);
    bool  sp   = (hpot >= th);
    S[(size_t)t * plane + i] = sp ? (_Float16)1.0f : (_Float16)0.0f;
    vm = sp ? 0.0f : hpot;
  }
}

// ---- fused GEMM (WMMA f16) + BN affine + LIF-over-T epilogue ----
// Block = 4 waves stacked along M: 128(Co) x 64(V) tile. The 64x32 spike tile
// is async-staged into double-buffered LDS and shared by all 4 waves; each
// wave runs 8 WMMAs per K-step (32x64 tile). Membrane state lives in VGPRs
// across the t-loop.
template <int LAYOUT>
__global__ __launch_bounds__(128) void gemm_bn_lif_kernel(
    const _Float16* __restrict__ W,    // (Co, Ci) f16 row-major
    const _Float16* __restrict__ XT,   // (T, B, V, Ci) f16 spikes, K-major
    const float* __restrict__ svec, const float* __restrict__ bvec,
    const float* __restrict__ convb,   // nullable conv bias
    _Float16* __restrict__ Out,        // spikes, layout-dependent
    float* __restrict__ OutV,          // optional f32 heads output (v spikes)
    int Co, int Ci, float th) {
  __shared__ __align__(16) _Float16 btile[2][64 * 32];

  const int tid  = threadIdx.x;
  const int wave = tid >> 5;
  const int lane = tid & 31;
  const int half = lane >> 4;
  const int lr   = lane & 15;
  const int n0   = blockIdx.x * 64;
  const int m0   = blockIdx.y * 128 + wave * 32;
  const int b    = blockIdx.z;
  const int KT_  = Ci / 32;
  const size_t planeX = (size_t)VV * Ci;

  float sc[2][8], bs[2][8];
#pragma unroll
  for (int mi2 = 0; mi2 < 2; ++mi2)
#pragma unroll
    for (int r = 0; r < 8; ++r) {
      int   c  = m0 + mi2 * 16 + half * 8 + r;
      float s_ = svec[c];
      sc[mi2][r] = s_;
      bs[mi2][r] = bvec[c] + (convb ? convb[c] * s_ : 0.0f);
    }

  float vm[2][4][8] = {};

  int p = 0;
  stage_btile(XT + (size_t)b * planeX, Ci, n0, 0, tid, 0);

  for (int t = 0; t < TT; ++t) {
    v8f acc[2][4] = {};
    for (int kt = 0; kt < KT_; ++kt) {
      const int k0 = kt * 32;
      wait_async0();
      __syncthreads();
      if (kt + 1 < KT_)
        stage_btile(XT + (size_t)(t * BB + b) * planeX, Ci, n0, k0 + 32, tid, p ^ 1);
      else if (t + 1 < TT)
        stage_btile(XT + (size_t)((t + 1) * BB + b) * planeX, Ci, n0, 0, tid, p ^ 1);
      v16h a0 = load_a_rowmajor(W, m0 + lr, Ci, k0, half);
      v16h a1 = load_a_rowmajor(W, m0 + 16 + lr, Ci, k0, half);
      const _Float16* bt = btile[p];
#pragma unroll
      for (int nt = 0; nt < 4; ++nt) {
        v16h bf = load_row16(bt + (nt * 16 + lr) * 32 + half * 16);
        acc[0][nt] = __builtin_amdgcn_wmma_f32_16x16x32_f16(false, a0, false, bf,
                                                            (short)0, acc[0][nt], false, false);
        acc[1][nt] = __builtin_amdgcn_wmma_f32_16x16x32_f16(false, a1, false, bf,
                                                            (short)0, acc[1][nt], false, false);
      }
      p ^= 1;
    }
    // epilogue: BN affine + LIF step, emit spikes
#pragma unroll
    for (int mi2 = 0; mi2 < 2; ++mi2)
#pragma unroll
      for (int nt = 0; nt < 4; ++nt)
#pragma unroll
        for (int r = 0; r < 8; ++r) {
          int   c    = m0 + mi2 * 16 + half * 8 + r;
          int   n    = n0 + nt * 16 + lr;
          float u    = acc[mi2][nt][r] * sc[mi2][r] + bs[mi2][r];
          float hpot = 0.5f * (vm[mi2][nt][r] + u);
          bool  sp   = (hpot >= th);
          vm[mi2][nt][r] = sp ? 0.0f : hpot;
          _Float16 sval = sp ? (_Float16)1.0f : (_Float16)0.0f;
          if (LAYOUT == LAY_PLAIN) {
            Out[(size_t)(t * BB + b) * VV * Co + (size_t)n * Co + c] = sval;
          } else {
            int    h    = c >> 6, j = c & 63;
            int    ncq  = t >> 1;
            int    l    = (t & 1) * VV + n;
            size_t base = ((size_t)(ncq * BB + b)) * HH + h;
            if (LAYOUT == LAY_QV)
              Out[(base * LL + l) * DD + j] = sval;            // (nc,B,H,L,d)
            else
              Out[(base * DD + j) * LL + l] = sval;            // (nc,B,H,d,L)
            if (OutV)                                          // v spikes -> output 1
              OutV[(((size_t)(t * BB + b) * HH + h) * VV + n) * DD + j] = sp ? 1.0f : 0.0f;
          }
        }
  }
}

// ---- GEMM + BN affine + residual add (proj and MLP-f2) ----
template <bool RES_CMAJOR, bool OUT_CMAJOR>
__global__ __launch_bounds__(128) void gemm_bn_res_kernel(
    const _Float16* __restrict__ W, const _Float16* __restrict__ XT, // (TB, V, Ci)
    const float* __restrict__ svec, const float* __restrict__ bvec,
    const float* __restrict__ convb,
    const float* __restrict__ Res, float* __restrict__ Out,
    int Co, int Ci) {
  __shared__ __align__(16) _Float16 btile[2][64 * 32];

  const int tid  = threadIdx.x;
  const int wave = tid >> 5;
  const int lane = tid & 31;
  const int half = lane >> 4;
  const int lr   = lane & 15;
  const int n0   = blockIdx.x * 64;
  const int m0   = blockIdx.y * 128 + wave * 32;
  const int tb   = blockIdx.z;
  const int KT_  = Ci / 32;

  const _Float16* XTtb = XT + (size_t)tb * VV * Ci;
  int p = 0;
  stage_btile(XTtb, Ci, n0, 0, tid, 0);

  v8f acc[2][4] = {};
  for (int kt = 0; kt < KT_; ++kt) {
    const int k0 = kt * 32;
    wait_async0();
    __syncthreads();
    if (kt + 1 < KT_) stage_btile(XTtb, Ci, n0, k0 + 32, tid, p ^ 1);
    v16h a0 = load_a_rowmajor(W, m0 + lr, Ci, k0, half);
    v16h a1 = load_a_rowmajor(W, m0 + 16 + lr, Ci, k0, half);
    const _Float16* bt = btile[p];
#pragma unroll
    for (int nt = 0; nt < 4; ++nt) {
      v16h bf = load_row16(bt + (nt * 16 + lr) * 32 + half * 16);
      acc[0][nt] = __builtin_amdgcn_wmma_f32_16x16x32_f16(false, a0, false, bf,
                                                          (short)0, acc[0][nt], false, false);
      acc[1][nt] = __builtin_amdgcn_wmma_f32_16x16x32_f16(false, a1, false, bf,
                                                          (short)0, acc[1][nt], false, false);
    }
    p ^= 1;
  }
#pragma unroll
  for (int mi2 = 0; mi2 < 2; ++mi2)
#pragma unroll
    for (int nt = 0; nt < 4; ++nt)
#pragma unroll
      for (int r = 0; r < 8; ++r) {
        int    c   = m0 + mi2 * 16 + half * 8 + r;
        int    n   = n0 + nt * 16 + lr;
        float  s_  = svec[c];
        float  u   = acc[mi2][nt][r] * s_ + bvec[c] + (convb ? convb[c] * s_ : 0.0f);
        size_t icm = ((size_t)tb * Co + c) * VV + n;  // C-major (T,B,C,V)
        size_t itp = ((size_t)tb * VV + n) * Co + c;  // K-major (T,B,V,C)
        float  res = RES_CMAJOR ? Res[icm] : Res[itp];
        if (OUT_CMAJOR) Out[icm] = u + res;
        else            Out[itp] = u + res;
      }
}

// ---- chunked linear attention: attn = (1/N) K^T V (64x64, via LDS), out = Q attn ----
__global__ __launch_bounds__(256) void attn_kernel(
    const _Float16* __restrict__ Q,    // (G, L, D)
    const _Float16* __restrict__ KT_,  // (G, D, L)
    const _Float16* __restrict__ VT,   // (G, D, L)
    float* __restrict__ Y) {           // (T, B, V, C)  K-major
  __shared__ __align__(16) _Float16 attn_sT[DD * DD];  // [nV][kK]
  const int g    = blockIdx.x;         // ((nc*B)+b)*H + h
  const int wave = threadIdx.x >> 5;
  const int lane = threadIdx.x & 31;
  const int half = lane >> 4;
  const int lr   = lane & 15;

  const _Float16* Qg = Q   + (size_t)g * LL * DD;
  const _Float16* Kg = KT_ + (size_t)g * DD * LL;
  const _Float16* Vg = VT  + (size_t)g * DD * LL;

  // phase 1: 64x64 K^T V -> 16 tiles (2 per wave), K-dim = L = 2048
#pragma unroll
  for (int tt0 = 0; tt0 < 2; ++tt0) {
    int tt = wave * 2 + tt0;
    int mi = tt >> 2, ni = tt & 3;
    v8f acc = {};
    for (int k0 = 0; k0 < LL; k0 += 32) {
      v16h a  = load_a_rowmajor(Kg, mi * 16 + lr, LL, k0, half);
      v16h bf = load_row16(Vg + (size_t)(ni * 16 + lr) * LL + k0 + half * 16);
      acc = __builtin_amdgcn_wmma_f32_16x16x32_f16(false, a, false, bf,
                                                   (short)0, acc, false, false);
    }
#pragma unroll
    for (int r = 0; r < 8; ++r)
      attn_sT[(ni * 16 + lr) * DD + mi * 16 + half * 8 + r] =
          (_Float16)(acc[r] * (1.0f / (float)VV));
  }
  __syncthreads();

  const int h   = g & 7;
  const int b   = (g >> 3) & 7;
  const int ncq = g >> 6;

  // phase 2: out = Q(2048x64) @ attn(64x64); Q fragments reused across 4 N-tiles
  for (int mi = wave; mi < 128; mi += 8) {
    v16h a0 = load_a_rowmajor(Qg, mi * 16 + lr, DD, 0, half);
    v16h a1 = load_a_rowmajor(Qg, mi * 16 + lr, DD, 32, half);
#pragma unroll
    for (int ni = 0; ni < 4; ++ni) {
      v16h b0 = load_row16(&attn_sT[(ni * 16 + lr) * DD + half * 16]);
      v16h b1 = load_row16(&attn_sT[(ni * 16 + lr) * DD + 32 + half * 16]);
      v8f acc = {};
      acc = __builtin_amdgcn_wmma_f32_16x16x32_f16(false, a0, false, b0,
                                                   (short)0, acc, false, false);
      acc = __builtin_amdgcn_wmma_f32_16x16x32_f16(false, a1, false, b1,
                                                   (short)0, acc, false, false);
#pragma unroll
      for (int r = 0; r < 8; ++r) {
        int l = mi * 16 + half * 8 + r;
        int j = ni * 16 + lr;
        int t = ncq * 2 + (l >> 10);
        int n = l & (VV - 1);
        int c = h * DD + j;
        Y[(size_t)(t * BB + b) * VV * CC + (size_t)n * CC + c] = acc[r];
      }
    }
  }
}

// ---- host orchestration ----

extern "C" void kernel_launch(void* const* d_in, const int* in_sizes, int n_in,
                              void* d_out, int out_size, void* d_ws, size_t ws_size,
                              hipStream_t stream) {
  (void)in_sizes; (void)n_in; (void)out_size; (void)ws_size;

  const float* x    = (const float*)d_in[0];
  const float* qw   = (const float*)d_in[1];
  const float* qs   = (const float*)d_in[2];
  const float* qb   = (const float*)d_in[3];
  const float* kw   = (const float*)d_in[4];
  const float* ks_  = (const float*)d_in[5];
  const float* kb   = (const float*)d_in[6];
  const float* vw   = (const float*)d_in[7];
  const float* vs   = (const float*)d_in[8];
  const float* vb   = (const float*)d_in[9];
  const float* pw   = (const float*)d_in[10];
  const float* pwb  = (const float*)d_in[11];
  const float* ps   = (const float*)d_in[12];
  const float* psb  = (const float*)d_in[13];
  const float* f1w  = (const float*)d_in[14];
  const float* f1b  = (const float*)d_in[15];
  const float* f1s  = (const float*)d_in[16];
  const float* f1sb = (const float*)d_in[17];
  const float* f2w  = (const float*)d_in[18];
  const float* f2b  = (const float*)d_in[19];
  const float* f2s  = (const float*)d_in[20];
  const float* f2sb = (const float*)d_in[21];

  const size_t N_TBCV = (size_t)TT * BB * CC * VV;  // 16,777,216
  float* out0 = (float*)d_out;
  float* out1 = out0 + N_TBCV;                       // v spikes (T,B,H,N,d)

  char* w = (char*)d_ws;
  auto take = [&](size_t bytes) { char* p = w; w += bytes; return p; };
  _Float16* Wq     = (_Float16*)take((size_t)CC * CC * 2);
  _Float16* Wk     = (_Float16*)take((size_t)CC * CC * 2);
  _Float16* Wv     = (_Float16*)take((size_t)CC * CC * 2);
  _Float16* Wp     = (_Float16*)take((size_t)CC * CC * 2);
  _Float16* Wf1    = (_Float16*)take((size_t)HID * CC * 2);
  _Float16* Wf2    = (_Float16*)take((size_t)CC * HID * 2);
  _Float16* xs16   = (_Float16*)take(N_TBCV * 2);   // (T,B,V,C)
  _Float16* q16    = (_Float16*)take(N_TBCV * 2);   // (nc,B,H,L,d)
  _Float16* kT16   = (_Float16*)take(N_TBCV * 2);   // (nc,B,H,d,L)
  _Float16* vT16   = (_Float16*)take(N_TBCV * 2);   // (nc,B,H,d,L)
  float*    y32    = (float*)   take(N_TBCV * 4);   // (T,B,V,C)
  _Float16* yl16   = (_Float16*)take(N_TBCV * 2);   // (T,B,V,C)
  float*    xat32  = (float*)   take(N_TBCV * 4);   // (T,B,V,C)
  _Float16* m1_16  = (_Float16*)take(N_TBCV * 2);   // (T,B,V,C)
  _Float16* hid16  = (_Float16*)take((size_t)TT * BB * HID * VV * 2); // (T,B,V,HID)

  auto cvt = [&](const float* src, _Float16* dst, int n) {
    cvt_f16_kernel<<<(n + 255) / 256, 256, 0, stream>>>(src, dst, n);
  };
  cvt(qw, Wq, CC * CC);  cvt(kw, Wk, CC * CC);  cvt(vw, Wv, CC * CC);
  cvt(pw, Wp, CC * CC);  cvt(f1w, Wf1, HID * CC);  cvt(f2w, Wf2, CC * HID);

  const size_t planeC = (size_t)BB * CC * VV;
  const int    lifBlocks = (int)((planeC + 255) / 256);

  // shortcut LIF (transposes x to K-major spikes)
  lif_x_tr_kernel<<<lifBlocks, 256, 0, stream>>>(x, xs16);

  // q/k/v: GEMM + BN + LIF fused over T
  dim3 blk(128);
  dim3 gqkv(VV / 64, CC / 128, BB);
  gemm_bn_lif_kernel<LAY_QV><<<gqkv, blk, 0, stream>>>(Wq, xs16, qs, qb, nullptr,
                                                       q16, nullptr, CC, CC, 1.0f);
  gemm_bn_lif_kernel<LAY_KT><<<gqkv, blk, 0, stream>>>(Wk, xs16, ks_, kb, nullptr,
                                                       kT16, nullptr, CC, CC, 1.0f);
  gemm_bn_lif_kernel<LAY_KT><<<gqkv, blk, 0, stream>>>(Wv, xs16, vs, vb, nullptr,
                                                       vT16, out1, CC, CC, 1.0f);

  // chunked attention
  attn_kernel<<<NC * BB * HH, 256, 0, stream>>>(q16, kT16, vT16, y32);

  // attn LIF (v_th = 0.5)
  lif_lin_kernel<<<lifBlocks, 256, 0, stream>>>(y32, yl16, planeC, 0.5f);

  // proj: GEMM + bias + BN + residual(x); out K-major
  dim3 gres(VV / 64, CC / 128, TT * BB);
  gemm_bn_res_kernel<true, false><<<gres, blk, 0, stream>>>(Wp, yl16, ps, psb, pwb,
                                                            x, xat32, CC, CC);

  // MLP
  lif_lin_kernel<<<lifBlocks, 256, 0, stream>>>(xat32, m1_16, planeC, 1.0f);
  dim3 gf1(VV / 64, HID / 128, BB);
  gemm_bn_lif_kernel<LAY_PLAIN><<<gf1, blk, 0, stream>>>(Wf1, m1_16, f1s, f1sb, f1b,
                                                         hid16, nullptr, HID, CC, 1.0f);
  gemm_bn_res_kernel<false, true><<<gres, blk, 0, stream>>>(Wf2, hid16, f2s, f2sb, f2b,
                                                            xat32, out0, CC, HID);
}
